// LanguageModelScratch_32435593020215
// MI455X (gfx1250) — matrix operand order repaired and checked by reading, not compiled, not run on hardware
//
#include <hip/hip_runtime.h>
#include <hip/hip_bf16.h>
#include <math.h>

// ---------------------------------------------------------------------------
// RNN language model on MI455X (gfx1250, wave32, WMMA).
//   B=64, T=256, V=4096, H=512
//   recurrence:  h_t = tanh(Wxh[x_t] + h_{t-1} @ Whh + bh)   (serial over T)
//   head:        out[b,t,:] = h_{t,b} @ Wo + Bo              (16384x4096x512 GEMM)
// bf16 operands through v_wmma_f32_16x16x32_bf16. Operands live in L2 (192MB);
// HBM traffic dominated by the 268MB fp32 output write (~11.5us at 23.3TB/s).
// The head GEMM is staged through LDS with async global->LDS copies to cut L2
// redundancy (~0.5GB total reads) and overlap copy with WMMA.
// ---------------------------------------------------------------------------

#define Bdim 64
#define Tdim 256
#define Vdim 4096
#define Hdim 512

typedef __bf16 v16bf __attribute__((ext_vector_type(16)));
typedef float  v8f   __attribute__((ext_vector_type(8)));

union FragU { v16bf v; uint4 q[2]; };

// A-fragment (16x32 bf16, M on lanes). Per ISA 16-bit A 16x32 table:
//   lanes 0-15 : M=lane,    halves 0..7 -> K=k0+0..7,  halves 8..15 -> K=k0+16..23
//   lanes16-31 : M=lane-16, halves 0..7 -> K=k0+8..15, halves 8..15 -> K=k0+24..31
__device__ __forceinline__ v16bf load_a_frag(const __hip_bfloat16* rowBase, int k0, int lane) {
    const int off = (lane >= 16) ? 8 : 0;
    FragU f;
    f.q[0] = *reinterpret_cast<const uint4*>(rowBase + k0 + off);
    f.q[1] = *reinterpret_cast<const uint4*>(rowBase + k0 + 16 + off);
    return f.v;
}

// B-fragment (32x16 bf16, N on lanes), same K order as the A-fragment:
//   lanes 0-15 : N=lane,    halves 0..7 -> K=k0+0..7,  halves 8..15 -> K=k0+16..23
//   lanes16-31 : N=lane-16, halves 0..7 -> K=k0+8..15, halves 8..15 -> K=k0+24..31
// rowBase is a K-contiguous row of the transposed weight (N-major).
__device__ __forceinline__ v16bf load_b_frag(const __hip_bfloat16* rowBase, int k0, int lane) {
    const int off = (lane >= 16) ? 8 : 0;
    FragU f;
    f.q[0] = *reinterpret_cast<const uint4*>(rowBase + k0 + off);
    f.q[1] = *reinterpret_cast<const uint4*>(rowBase + k0 + 16 + off);
    return f.v;
}

__device__ __forceinline__ v8f wmma_bf16(v16bf a, v16bf b, v8f c) {
    return __builtin_amdgcn_wmma_f32_16x16x32_bf16(false, a, false, b,
                                                   (short)0, c, false, false);
}

// Async global->LDS 16B copy (gfx1250, tracked by ASYNCcnt).
__device__ __forceinline__ void async_b128(const __hip_bfloat16* g, __hip_bfloat16* l) {
    unsigned      lds = (unsigned)(uintptr_t)l;
    unsigned long long ga = (unsigned long long)(uintptr_t)g;
    asm volatile("global_load_async_to_lds_b128 %0, %1, off"
                 :: "v"(lds), "v"(ga) : "memory");
}
__device__ __forceinline__ void wait_async0() {
    asm volatile("s_wait_asynccnt 0x0" ::: "memory");
}

// ---------------------------------------------------------------------------
// Prep: bf16 transposed weights. WhhT[n*H+k] = Whh[k,n]; WoT[v*H+k] = Wo[k,v].
// ---------------------------------------------------------------------------
__global__ void __launch_bounds__(256) prep_kernel(const float* __restrict__ Whh,
                                                   const float* __restrict__ Wo,
                                                   __hip_bfloat16* __restrict__ WhhT,
                                                   __hip_bfloat16* __restrict__ WoT) {
    const int i = blockIdx.x * blockDim.x + threadIdx.x;
    if (i < Hdim * Hdim) {
        const int n = i / Hdim, k = i % Hdim;
        WhhT[i] = __float2bfloat16(Whh[k * Hdim + n]);
    }
    if (i < Vdim * Hdim) {
        const int v = i / Hdim, k = i % Hdim;
        WoT[i] = __float2bfloat16(Wo[(size_t)k * Vdim + v]);
    }
}

// ---------------------------------------------------------------------------
// Recurrence: persistent kernel, 16 blocks x 256 threads (128 waves).
// Each wave owns one 16x16 tile of the 64x512 state; its Whh slice (16 cols x
// K=512) is preloaded once into 16 B-fragments (128 VGPRs). Per step: token /
// Wxh gathers issued first (hidden under GEMM), A-fragments software-pipelined
// 2 deep so loads overlap WMMAs, then tanh epilogue + global grid barrier.
// ---------------------------------------------------------------------------
__global__ void __launch_bounds__(256) rnn_kernel(const int*   __restrict__ x,
                                                  const float* __restrict__ Wxh,
                                                  const float* __restrict__ bh,
                                                  const __hip_bfloat16* __restrict__ WhhT,
                                                  const __hip_bfloat16* __restrict__ h0,
                                                  __hip_bfloat16* __restrict__ hs,
                                                  unsigned* __restrict__ ctr) {
    const int lane = threadIdx.x & 31;
    const int wave = threadIdx.x >> 5;
    const int tile = blockIdx.x * 8 + wave;   // 0..127
    const int mt   = tile >> 5;               // 0..3  : 16-row tile of batch dim
    const int nt   = tile & 31;               // 0..31 : 16-col tile of H dim

    const int n    = nt * 16 + (lane & 15);   // output column owned by this lane
    const float bhn = bh[n];

    v16bf bfrag[16];
    #pragma unroll
    for (int kk = 0; kk < 16; ++kk)
        bfrag[kk] = load_b_frag(WhhT + (size_t)n * Hdim, kk * 32, lane);

    const int rowA = mt * 16 + (lane & 15);            // A row this lane loads
    const int rowE = mt * 16 + ((lane >= 16) ? 8 : 0); // first C row this lane owns

    for (int t = 0; t < Tdim; ++t) {
        const __hip_bfloat16* aRow = (t == 0)
            ? (h0 + (size_t)rowA * Hdim)
            : (hs + ((size_t)rowA * Tdim + (t - 1)) * Hdim);

        // Issue the epilogue gathers early; they are independent of the GEMM.
        float xv[8];
        #pragma unroll
        for (int e = 0; e < 8; ++e) {
            const int tok = x[(rowE + e) * Tdim + t];
            xv[e] = Wxh[(size_t)tok * Hdim + n];
        }

        // Software-pipelined A stream: keep 2 fragments in flight.
        v16bf a0 = load_a_frag(aRow, 0, lane);
        v16bf a1 = load_a_frag(aRow, 32, lane);
        v8f acc = {};
        #pragma unroll
        for (int kk = 0; kk < 16; ++kk) {
            v16bf a2 = a1;
            if (kk + 2 < 16) a2 = load_a_frag(aRow, (kk + 2) * 32, lane);
            acc = wmma_bf16(a0, bfrag[kk], acc);
            a0 = a1; a1 = a2;
        }

        #pragma unroll
        for (int e = 0; e < 8; ++e) {
            const int b = rowE + e;
            float v = acc[e] + xv[e] + bhn;
            v = tanhf(v);
            hs[((size_t)b * Tdim + t) * Hdim + n] = __float2bfloat16(v);
        }

        // Grid-wide barrier: step t fully written before anyone reads it at t+1.
        __syncthreads();
        if (threadIdx.x == 0) {
            __threadfence();
            atomicAdd(ctr, 1u);
            const unsigned target = gridDim.x * (unsigned)(t + 1);
            while (__hip_atomic_load(ctr, __ATOMIC_ACQUIRE, __HIP_MEMORY_SCOPE_AGENT) < target) {
                __builtin_amdgcn_s_sleep(1);
            }
            __threadfence();
        }
        __syncthreads();
    }
}

// ---------------------------------------------------------------------------
// Output head: C[16384,4096] = hs(bf16) * WoT(bf16)^T + Bo.
// 256x256 block tile, 512 threads (16 waves, 4x4 grid of 64x64 wave tiles).
// K chunks of 32 staged into double-buffered LDS via async global->LDS b128
// copies (ASYNCcnt); WMMAs consume the previous chunk while the next streams.
// LDS rows padded to 40 halves (80B, 16B-aligned) to spread banks.
// ---------------------------------------------------------------------------
#define LROW 40                      // padded row length in halves
#define LTILE (256 * LROW)           // one 256x32 tile (padded)

__global__ void __launch_bounds__(512) outproj_kernel(const __hip_bfloat16* __restrict__ hs,
                                                      const __hip_bfloat16* __restrict__ WoT,
                                                      const float* __restrict__ Bo,
                                                      float* __restrict__ out) {
    __shared__ __hip_bfloat16 ldsA[2][LTILE];
    __shared__ __hip_bfloat16 ldsB[2][LTILE];

    const int tid  = threadIdx.x;
    const int lane = tid & 31;
    const int wave = tid >> 5;
    const int wm   = wave & 3;             // 0..3
    const int wn   = wave >> 2;            // 0..3
    const int Mblk = blockIdx.y * 256;
    const int Nblk = blockIdx.x * 256;

    const __hip_bfloat16* gA = hs  + (size_t)Mblk * Hdim;   // 256 rows, K-major
    const __hip_bfloat16* gB = WoT + (size_t)Nblk * Hdim;   // 256 rows, K-major

    // Stage one 256x32 chunk of A and B into LDS buffer `buf` (4 async b128/thread).
    auto stage = [&](int buf, int k0) {
        #pragma unroll
        for (int s = 0; s < 2; ++s) {          // A: 1024 16B units / 512 threads
            const int u = s * 512 + tid;
            const int r = u >> 2, c = (u & 3) * 8;
            async_b128(gA + (size_t)r * Hdim + k0 + c, &ldsA[buf][r * LROW + c]);
        }
        #pragma unroll
        for (int s = 0; s < 2; ++s) {          // B
            const int u = s * 512 + tid;
            const int r = u >> 2, c = (u & 3) * 8;
            async_b128(gB + (size_t)r * Hdim + k0 + c, &ldsB[buf][r * LROW + c]);
        }
    };

    v8f acc[4][4] = {};

    stage(0, 0);
    wait_async0();
    __syncthreads();

    for (int kk = 0; kk < 16; ++kk) {
        const int cur = kk & 1;
        if (kk + 1 < 16) stage(1 - cur, (kk + 1) * 32);

        // Consume current chunk from LDS.
        v16bf af[4];
        #pragma unroll
        for (int i = 0; i < 4; ++i)
            af[i] = load_a_frag(&ldsA[cur][(wm * 64 + i * 16 + (lane & 15)) * LROW], 0, lane);
        #pragma unroll
        for (int j = 0; j < 4; ++j) {
            v16bf bf = load_b_frag(&ldsB[cur][(wn * 64 + j * 16 + (lane & 15)) * LROW], 0, lane);
            #pragma unroll
            for (int i = 0; i < 4; ++i)
                acc[i][j] = wmma_bf16(af[i], bf, acc[i][j]);
        }

        if (kk + 1 < 16) {
            wait_async0();      // my async copies done (others wait on theirs)
            __syncthreads();    // everyone's copies + reads done
        }
    }

    const int rowE = Mblk + wm * 64 + ((lane >= 16) ? 8 : 0);
    float bo[4];
    #pragma unroll
    for (int j = 0; j < 4; ++j) bo[j] = Bo[Nblk + wn * 64 + j * 16 + (lane & 15)];

    #pragma unroll
    for (int i = 0; i < 4; ++i) {
        #pragma unroll
        for (int j = 0; j < 4; ++j) {
            const int col = Nblk + wn * 64 + j * 16 + (lane & 15);
            #pragma unroll
            for (int e = 0; e < 8; ++e) {
                const int r = rowE + i * 16 + e;
                out[(size_t)r * Vdim + col] = acc[i][j][e] + bo[j];
            }
        }
    }
}

// ---------------------------------------------------------------------------
// Workspace layout (bytes):
//   [0, 256)        : grid-barrier counter (memset 0 each launch)
//   [256, +64K)     : h0 zero state, 64x512 bf16 (memset each launch)
//   then WhhT 512x512 bf16 (512 KB), WoT 4096x512 bf16 (4 MB),
//   then hs 64x256x512 bf16 (16 MB).  Total ~21.6 MB.
// ---------------------------------------------------------------------------
extern "C" void kernel_launch(void* const* d_in, const int* in_sizes, int n_in,
                              void* d_out, int out_size, void* d_ws, size_t ws_size,
                              hipStream_t stream) {
    (void)in_sizes; (void)n_in; (void)out_size; (void)ws_size;
    const int*   x   = (const int*)  d_in[0];
    const float* Wxh = (const float*)d_in[1];
    const float* Whh = (const float*)d_in[2];
    const float* bh  = (const float*)d_in[3];
    const float* Wo  = (const float*)d_in[4];
    const float* Bo  = (const float*)d_in[5];
    float* out = (float*)d_out;

    char* ws = (char*)d_ws;
    unsigned*        ctr  = (unsigned*)ws;
    __hip_bfloat16*  h0   = (__hip_bfloat16*)(ws + 256);
    __hip_bfloat16*  WhhT = (__hip_bfloat16*)(ws + 256 + 65536);
    __hip_bfloat16*  WoT  = (__hip_bfloat16*)(ws + 256 + 65536 + Hdim * Hdim * 2);
    __hip_bfloat16*  hs   = (__hip_bfloat16*)(ws + 256 + 65536 + Hdim * Hdim * 2
                                                 + (size_t)Vdim * Hdim * 2);

    hipMemsetAsync(ws, 0, 256 + 65536, stream);

    const int prepN = Vdim * Hdim;  // covers the smaller Whh transpose too
    prep_kernel<<<(prepN + 255) / 256, 256, 0, stream>>>(Whh, Wo, WhhT, WoT);
    rnn_kernel<<<16, 256, 0, stream>>>(x, Wxh, bh, WhhT, h0, hs, ctr);
    outproj_kernel<<<dim3(Vdim / 256, (Bdim * Tdim) / 256), 512, 0, stream>>>(hs, WoT, Bo, out);
}
